// CausalSTFT_88510686036469
// MI455X (gfx1250) — matrix operand order repaired and checked by reading, not compiled, not run on hardware
//
#include <hip/hip_runtime.h>
#include <stdint.h>

// Causal STFT magnitude as a bf16 WMMA GEMM for gfx1250 (MI455X).
// M = 16*2048 frames, K = 1024, N = 513 cos + 513 sin, epilogue sqrt(c^2+s^2).
// Compute-bound (~69 GFLOP); x (33.5MB) + W (4.2MB) are L2-resident (192MB L2).
//
// Fast path (ws_size permitting): pre-convert x (with 1023-sample causal pad)
// and W (with zero guard rows) to bf16 in d_ws, then double-buffer LDS slabs
// staged with global_load_async_to_lds_b128 (ASYNCcnt): asyncs for slab kk+1
// are issued before computing slab kk, so the DMA runs under the WMMAs.
// Fallback: branchless inline f32->bf16 staging, single buffer.

#define N_FFT   1024
#define HOP     256
#define NFREQ   513            // N_FFT/2 + 1
#define NT      2048           // frames per batch
#define NB      16
#define NSAMP   (HOP * NT)     // 524288
#define MT      256            // frames per workgroup (8 waves x 2 subtiles x 16)
#define KS      64             // K slab per iteration
#define NKK     (N_FFT / KS)   // 16 K-steps
#define ROWU    36             // dwords per LDS row (72 bf16): 144B, 16B-aligned, bank-perfect
#define XROW    (NSAMP + 1032) // padded bf16 x row: 1023 zeros + samples + slack (mult of 8)
#define SINB    528            // sin rows base in padded bf16 weight
#define WROWS   1056           // cos[0,513) | zeros | sin[528,1041) | zeros

typedef __attribute__((ext_vector_type(2)))  __bf16 v2bf;
typedef __attribute__((ext_vector_type(8)))  float  v8f;
typedef __attribute__((ext_vector_type(8)))  __bf16 v8bf;
typedef __attribute__((ext_vector_type(16))) __bf16 v16bf;

__device__ __forceinline__ unsigned pack2_bf16(float a, float b) {
    v2bf p = { (__bf16)a, (__bf16)b };
    return __builtin_bit_cast(unsigned, p);
}

// ---- one-time f32 -> bf16 converts into d_ws ------------------------------

__global__ __launch_bounds__(256)
void stft_cvt_x_pad(const float* __restrict__ x, __bf16* __restrict__ xp) {
    size_t q = (size_t)blockIdx.x * 256 + threadIdx.x;     // packed-pair index
    if (q >= (size_t)NB * (XROW / 2)) return;
    int b = (int)(q / (XROW / 2));
    int j = (int)(q % (XROW / 2)) * 2;                     // element in padded row
    float a0 = 0.0f, a1 = 0.0f;
    int i0 = j - 1023, i1 = j - 1022;
    if (i0 >= 0 && i0 < NSAMP) a0 = x[(size_t)b * NSAMP + i0];
    if (i1 >= 0 && i1 < NSAMP) a1 = x[(size_t)b * NSAMP + i1];
    *(unsigned*)(xp + (size_t)b * XROW + j) = pack2_bf16(a0, a1);
}

__global__ __launch_bounds__(256)
void stft_cvt_w(const float* __restrict__ w, __bf16* __restrict__ wb) {
    int q = blockIdx.x * 256 + threadIdx.x;                // packed-pair index
    if (q >= WROWS * (N_FFT / 2)) return;
    int row = q >> 9;
    int k   = (q & 511) * 2;
    int src = -1;
    if (row < NFREQ) src = row;                            // cos rows
    else if (row >= SINB && row < SINB + NFREQ) src = NFREQ + (row - SINB);
    float a0 = 0.0f, a1 = 0.0f;
    if (src >= 0) {
        const float* p = w + (size_t)src * N_FFT + k;
        a0 = p[0]; a1 = p[1];
    }
    *(unsigned*)(wb + (size_t)row * N_FFT + k) = pack2_bf16(a0, a1);
}

// ---- main kernel ----------------------------------------------------------

template <bool PRE>
__global__ __launch_bounds__(256)
void stft_mag_wmma(const float* __restrict__ x, const float* __restrict__ w,
                   const __bf16* __restrict__ xp, const __bf16* __restrict__ wb,
                   float* __restrict__ out)
{
    // double-buffered A/W slabs (fallback uses buffer 0 only)
    __shared__ __align__(16) unsigned As32[2][MT][ROWU];  // 256 frames x 64 K bf16
    __shared__ __align__(16) unsigned Ws32[2][32][ROWU];  // 16 cos + 16 sin rows x 64 K
    __shared__ float MagS[16][257];

    const int tid = threadIdx.x;
    const int f0  = blockIdx.x * 16;
    const int t0  = blockIdx.y * MT;
    const int b   = blockIdx.z;

    const int lane = tid & 31;
    const int wv   = tid >> 5;
    const int mb0  = wv * 32;          // each wave owns 2 M-subtiles of 16
    const int l16  = lane & 15;
    const int lh   = lane >> 4;

    // LDS byte offsets for async destinations (generic LDS addr low 32 bits)
    const unsigned as_base = (unsigned)(uintptr_t)&As32[0][0][0];
    const unsigned ws_base = (unsigned)(uintptr_t)&Ws32[0][0][0];

    const float*  xb  = x  + (size_t)b * NSAMP;
    const __bf16* xbp = xp + (size_t)b * XROW;

    v8f accC0 = {}, accS0 = {}, accC1 = {}, accS1 = {};

    // issue one K-slab's async DMA batch (8 A chunks + 1 W chunk per thread)
    auto stage_async = [&](int kk) {
        const int k0  = kk * KS;
        const int buf = kk & 1;
        const unsigned abase = as_base + buf * (MT * ROWU * 4);
        const unsigned wbase = ws_base + buf * (32 * ROWU * 4);
        #pragma unroll
        for (int i = 0; i < 8; ++i) {
            int c  = i * 256 + tid;
            int m  = c >> 3;
            int kc = (c & 7) * 8;
            const __bf16* src = xbp + (size_t)(t0 + m) * HOP + k0 + kc;
            unsigned dst = abase + m * (ROWU * 4) + kc * 2;
            asm volatile("global_load_async_to_lds_b128 %0, %1, off"
                         :: "v"(dst), "v"(src) : "memory");
        }
        {
            int r  = tid >> 3;
            int kc = (tid & 7) * 8;
            int row = (r < 16) ? (f0 + (r & 15)) : (SINB + f0 + (r & 15));
            const __bf16* src = wb + (size_t)row * N_FFT + k0 + kc;
            unsigned dst = wbase + r * (ROWU * 4) + kc * 2;
            asm volatile("global_load_async_to_lds_b128 %0, %1, off"
                         :: "v"(dst), "v"(src) : "memory");
        }
    };

    // branchless inline-convert staging into buffer 0 (fallback path)
    auto stage_cvt = [&](int kk) {
        const int k0 = kk * KS;
        #pragma unroll
        for (int i = 0; i < 32; ++i) {
            int q2 = i * 256 + tid;
            int m  = q2 >> 5;
            int kp = q2 & 31;
            int s  = (t0 + m) * HOP + k0 + 2 * kp - (N_FFT - 1);
            int s0 = s < 0 ? 0 : s;
            int s1 = (s + 1) < 0 ? 0 : (s + 1);
            float a0 = xb[s0];
            float a1 = xb[s1];
            a0 = (s     >= 0) ? a0 : 0.0f;
            a1 = (s + 1 >= 0) ? a1 : 0.0f;
            As32[0][m][kp] = pack2_bf16(a0, a1);
        }
        #pragma unroll
        for (int i = 0; i < 4; ++i) {
            int q2 = i * 256 + tid;
            int r  = q2 >> 5;
            int kp = q2 & 31;
            int f  = f0 + (r & 15);
            int row = (r < 16) ? f : (NFREQ + f);
            row = row > 1025 ? 1025 : row;                 // clamp, select below
            const float* wp = w + (size_t)row * N_FFT + k0 + 2 * kp;
            float w0 = wp[0], w1 = wp[1];
            bool ok = (f < NFREQ);
            w0 = ok ? w0 : 0.0f;
            w1 = ok ? w1 : 0.0f;
            Ws32[0][r][kp] = pack2_bf16(w0, w1);
        }
    };

    // one K-slab of WMMAs from LDS buffer `buf`
    auto compute = [&](int buf) {
        const __bf16* arow0 = (const __bf16*)&As32[buf][mb0 + l16][0];
        const __bf16* arow1 = (const __bf16*)&As32[buf][mb0 + 16 + l16][0];
        const __bf16* bcrow = (const __bf16*)&Ws32[buf][l16][0];
        const __bf16* bsrow = (const __bf16*)&Ws32[buf][16 + l16][0];
        #pragma unroll
        for (int ks = 0; ks < KS; ks += 32) {
            // A 16x32 bf16: lanes 0-15 K {0..7,16..23}; lanes 16-31 K {8..15,24..31}
            v8bf a0lo = *(const v8bf*)(arow0 + ks + lh * 8);
            v8bf a0hi = *(const v8bf*)(arow0 + ks + 16 + lh * 8);
            v8bf a1lo = *(const v8bf*)(arow1 + ks + lh * 8);
            v8bf a1hi = *(const v8bf*)(arow1 + ks + 16 + lh * 8);
            v16bf a0 = __builtin_shufflevector(a0lo, a0hi,
                         0,1,2,3,4,5,6,7,8,9,10,11,12,13,14,15);
            v16bf a1 = __builtin_shufflevector(a1lo, a1hi,
                         0,1,2,3,4,5,6,7,8,9,10,11,12,13,14,15);
            // B 32x16 bf16: lanes 0-15 K {0..7,16..23} of col N=lane; lanes 16-31 K {8..15,24..31}
            v8bf bclo = *(const v8bf*)(bcrow + ks + lh * 8);
            v8bf bchi = *(const v8bf*)(bcrow + ks + 16 + lh * 8);
            v8bf bslo = *(const v8bf*)(bsrow + ks + lh * 8);
            v8bf bshi = *(const v8bf*)(bsrow + ks + 16 + lh * 8);
            v16bf bc = __builtin_shufflevector(bclo, bchi,
                         0,1,2,3,4,5,6,7,8,9,10,11,12,13,14,15);
            v16bf bs = __builtin_shufflevector(bslo, bshi,
                         0,1,2,3,4,5,6,7,8,9,10,11,12,13,14,15);
            accC0 = __builtin_amdgcn_wmma_f32_16x16x32_bf16(
                        false, a0, false, bc, (short)0, accC0, false, false);
            accS0 = __builtin_amdgcn_wmma_f32_16x16x32_bf16(
                        false, a0, false, bs, (short)0, accS0, false, false);
            accC1 = __builtin_amdgcn_wmma_f32_16x16x32_bf16(
                        false, a1, false, bc, (short)0, accC1, false, false);
            accS1 = __builtin_amdgcn_wmma_f32_16x16x32_bf16(
                        false, a1, false, bs, (short)0, accS1, false, false);
        }
    };

    if constexpr (PRE) {
        // software pipeline: DMA for slab kk+1 flies under compute of slab kk
        stage_async(0);
        for (int kk = 0; kk < NKK; ++kk) {
            asm volatile("s_wait_asynccnt 0x0" ::: "memory");
            __syncthreads();                  // slab kk visible; compute(kk-1) done by all
            if (kk + 1 < NKK) stage_async(kk + 1);
            compute(kk & 1);
        }
    } else {
        for (int kk = 0; kk < NKK; ++kk) {
            stage_cvt(kk);
            __syncthreads();
            compute(0);
            __syncthreads();
        }
    }

    // ---- epilogue: magnitude, transpose via LDS for coalesced stores
    // C layout: VGPR r -> frame = mb + r + 8*lh, freq = l16
    #pragma unroll
    for (int sub = 0; sub < 2; ++sub) {
        v8f aC = sub ? accC1 : accC0;
        v8f aS = sub ? accS1 : accS0;
        #pragma unroll
        for (int r = 0; r < 8; ++r) {
            float cc = aC[r], ss = aS[r];
            MagS[l16][mb0 + sub * 16 + r + lh * 8] =
                __builtin_amdgcn_sqrtf(fmaxf(cc * cc + ss * ss, 1e-12f));
        }
    }
    __syncthreads();

    #pragma unroll
    for (int i = 0; i < 16; ++i) {
        int q  = i * 256 + tid;
        int fl = q >> 8;       // local freq 0..15
        int tl = q & 255;      // local frame 0..255
        int f  = f0 + fl;
        if (f < NFREQ)
            out[((size_t)b * NFREQ + f) * NT + t0 + tl] = MagS[fl][tl];
    }
}

extern "C" void kernel_launch(void* const* d_in, const int* in_sizes, int n_in,
                              void* d_out, int out_size, void* d_ws, size_t ws_size,
                              hipStream_t stream) {
    (void)in_sizes; (void)n_in; (void)out_size;
    const float* x = (const float*)d_in[0];   // (16, 524288) f32
    const float* w = (const float*)d_in[1];   // (1026, 1, 1024) f32
    float* out = (float*)d_out;               // (16, 513, 2048) f32

    const size_t xbytes = (size_t)NB * XROW * sizeof(__bf16);
    const size_t wbytes = (size_t)WROWS * N_FFT * sizeof(__bf16);
    dim3 grid((NFREQ + 15) / 16, NT / MT, NB);   // 33 x 8 x 16

    if (ws_size >= xbytes + wbytes) {
        __bf16* xp = (__bf16*)d_ws;
        __bf16* wb = xp + (size_t)NB * XROW;
        size_t xpairs = (size_t)NB * (XROW / 2);
        stft_cvt_x_pad<<<(unsigned)((xpairs + 255) / 256), 256, 0, stream>>>(x, xp);
        int wpairs = WROWS * (N_FFT / 2);
        stft_cvt_w<<<(wpairs + 255) / 256, 256, 0, stream>>>(w, wb);
        stft_mag_wmma<true><<<grid, 256, 0, stream>>>(x, w, xp, wb, out);
    } else {
        stft_mag_wmma<false><<<grid, 256, 0, stream>>>(x, w, nullptr, nullptr, out);
    }
}
